// resBLSTM_27384711479502
// MI455X (gfx1250) — matrix operand order repaired and checked by reading, not compiled, not run on hardware
//
#include <hip/hip_runtime.h>
#include <hip/hip_bf16.h>

// ---------------- CDNA5 WMMA types ----------------
typedef __attribute__((ext_vector_type(16))) _Float16 v16h;
typedef __attribute__((ext_vector_type(8)))  _Float16 v8h;
typedef __attribute__((ext_vector_type(8)))  float    v8f;

// ---------------- problem constants ----------------
constexpr int H   = 161;        // hidden
constexpr int B   = 32;
constexpr int T   = 1000;
constexpr int G4  = 644;        // 4H
constexpr int NP  = 656;        // 4H padded to 41*16
constexpr int NT_LSTM = 41;
constexpr int KS_LSTM = 6;      // K=161 padded to 192 = 6*32
constexpr int KP  = 192;
constexpr int HP  = 176;        // H padded to 11*16
constexpr int NT_LIN = 11;
constexpr int KS_LIN = 11;      // 2H=322 -> two 176 halves -> 352 = 11*32
constexpr int K2P = 352;
constexpr long BT = (long)B * T;   // 32000 rows, multiple of 16

// packed B-fragment sizes (f16 elements)
constexpr size_t PK_LSTM = (size_t)NT_LSTM * KS_LSTM * 32 * 16;  // 125952
constexpr size_t PK_LIN  = (size_t)NT_LIN  * KS_LIN  * 32 * 16;  // 61952

// ---------------- workspace layout (bytes) ----------------
constexpr size_t SZ_PK   = PK_LSTM * 2;               // 251904
constexpr size_t OFF_WIH = 0;                         // [blk][dir] x6
constexpr size_t OFF_WHH = OFF_WIH + 6 * SZ_PK;
constexpr size_t SZ_BS   = (size_t)NP * 4;            // 2624
constexpr size_t OFF_BS  = OFF_WHH + 6 * SZ_PK;       // [blk][dir] x6
constexpr size_t SZ_WT   = PK_LIN * 2;                // 123904
constexpr size_t OFF_WT  = OFF_BS + 6 * SZ_BS;        // [blk] x3
constexpr size_t SZ_PRE  = (size_t)BT * NP * 2;       // 41,984,000
constexpr size_t OFF_PREF= OFF_WT + 3 * SZ_WT;
constexpr size_t OFF_PREB= OFF_PREF + SZ_PRE;
constexpr size_t SZ_H16  = (size_t)BT * HP * 2;       // 11,264,000
constexpr size_t OFF_HF  = OFF_PREB + SZ_PRE;
constexpr size_t OFF_HB  = OFF_HF + SZ_H16;           // hF,hB adjacent
constexpr size_t SZ_IO   = (size_t)BT * H * 4;        // 20,608,000
constexpr size_t OFF_IOA = OFF_HB + SZ_H16;
constexpr size_t OFF_IOB = OFF_IOA + SZ_IO;
// total ~151 MB

// ---------------- fragment helpers ----------------
// A-matrix 16x32 f16 fragment from an LDS/row-major tile (ldk f16 per row).
// Lanes 0-15: rows M=lane, kb=0 ; lanes 16-31: rows M=lane-16, kb=8.
// VGPR0-3 hold K = kb+0..7, VGPR4-7 hold K = kb+16..23.
__device__ __forceinline__ v16h frag_a(const _Float16* tile, int ldk, int lane) {
    int m  = lane & 15;
    int kb = (lane & 16) ? 8 : 0;
    const _Float16* r = tile + m * ldk + kb;
    v8h lo = *(const v8h*)(r);        // K = kb .. kb+7
    v8h hi = *(const v8h*)(r + 16);   // K = kb+16 .. kb+23
    return __builtin_shufflevector(lo, hi, 0,1,2,3,4,5,6,7,8,9,10,11,12,13,14,15);
}

// B fragment: pre-packed, one 32-byte vector load per lane.
__device__ __forceinline__ v16h frag_b_packed(const _Float16* pk, int nt, int ks, int KS, int lane) {
    return *((const v16h*)pk + ((size_t)(nt * KS + ks) * 32 + lane));
}

__device__ __forceinline__ v8f wmma_f16(v16h a, v16h b, v8f c) {
    return __builtin_amdgcn_wmma_f32_16x16x32_f16(false, a, false, b, (short)0, c, false, false);
}

__device__ __forceinline__ float sigm(float x) { return 1.0f / (1.0f + __expf(-x)); }
// tanh via sigmoid: avoids an ocml libcall on the serial path (v_exp_f32 based)
__device__ __forceinline__ float tanh_fast(float x) { return 2.0f * sigm(2.0f * x) - 1.0f; }

// ---------------- prep kernels ----------------
// Pack w[N][K] row-major into WMMA B-fragment order for tiles (NT x KS), zero-padded.
__global__ void k_pack_b(const float* __restrict__ w, int N, int K,
                         _Float16* __restrict__ pk, int NT, int KS) {
    int idx = blockIdx.x * blockDim.x + threadIdx.x;
    int total = NT * KS * 512;
    if (idx >= total) return;
    int e    = idx & 15;
    int lane = (idx >> 4) & 31;
    int t    = idx >> 9;
    int ks   = t % KS;
    int nt   = t / KS;
    int n = nt * 16 + (lane & 15);
    int k = ks * 32 + ((e < 8) ? e : e + 8) + ((lane & 16) ? 8 : 0);
    float v = (n < N && k < K) ? w[(size_t)n * K + k] : 0.0f;
    pk[idx] = (_Float16)v;
}

// Pack linear W [H][2H]: A's K-axis is [hF pad176 | hB pad176] -> map k->W column.
__global__ void k_pack_b_lin(const float* __restrict__ W, _Float16* __restrict__ pk) {
    int idx = blockIdx.x * blockDim.x + threadIdx.x;
    int total = NT_LIN * KS_LIN * 512;
    if (idx >= total) return;
    int e    = idx & 15;
    int lane = (idx >> 4) & 31;
    int t    = idx >> 9;
    int ks   = t % KS_LIN;
    int nt   = t / KS_LIN;
    int n = nt * 16 + (lane & 15);
    int k = ks * 32 + ((e < 8) ? e : e + 8) + ((lane & 16) ? 8 : 0);
    int seg = (k < HP) ? k : (k - HP);
    int col = ((k < HP) ? 0 : H) + seg;
    float v = (n < H && seg < H) ? W[(size_t)n * (2 * H) + col] : 0.0f;
    pk[idx] = (_Float16)v;
}

__global__ void k_bias_sum(const float* __restrict__ bih, const float* __restrict__ bhh,
                           float* __restrict__ bs) {
    int n = blockIdx.x * blockDim.x + threadIdx.x;
    if (n < NP) bs[n] = (n < G4) ? (bih[n] + bhh[n]) : 0.0f;
}

__global__ void k_zero32(unsigned int* __restrict__ p, long nwords) {
    long i = (long)blockIdx.x * blockDim.x + threadIdx.x;
    if (i < nwords) p[i] = 0u;
}

// ---------------- pre-activation GEMM ----------------
// pre[r][n] = sum_k x[r][k]*w_ih[n][k] + (b_ih+b_hh)[n], both directions. r = b*T+t.
__global__ void k_pregemm(const float* __restrict__ xin,
                          const _Float16* __restrict__ pkF, const _Float16* __restrict__ pkB,
                          const float* __restrict__ bsF, const float* __restrict__ bsB,
                          _Float16* __restrict__ preF, _Float16* __restrict__ preB) {
    __shared__ _Float16 Atile[16 * KP];   // 6144 B
    const int tid = threadIdx.x;          // 256 threads = 8 waves
    const long m0 = (long)blockIdx.x * 16;
    for (int e = tid; e < 16 * KP; e += 256) {
        int r = e / KP, k = e - r * KP;
        float v = (k < H) ? xin[(m0 + r) * H + k] : 0.0f;
        Atile[e] = (_Float16)v;
    }
    __syncthreads();
    const int wave = tid >> 5, lane = tid & 31;
    for (int job = wave; job < 2 * NT_LSTM; job += 8) {
        const int dir = job / NT_LSTM;
        const int nt  = job - dir * NT_LSTM;
        const _Float16* pk = dir ? pkB : pkF;
        const float*    bs = dir ? bsB : bsF;
        _Float16*      pre = dir ? preB : preF;
        const int n = nt * 16 + (lane & 15);
        const float bias = bs[n];
        // batch-preload all B fragments -> one clause of loads
        v16h bf[KS_LSTM];
        #pragma unroll
        for (int ks = 0; ks < KS_LSTM; ++ks)
            bf[ks] = frag_b_packed(pk, nt, ks, KS_LSTM, lane);
        v8f acc = {bias, bias, bias, bias, bias, bias, bias, bias};
        #pragma unroll
        for (int ks = 0; ks < KS_LSTM; ++ks) {
            v16h a = frag_a(Atile + ks * 32, KP, lane);
            acc = wmma_f16(a, bf[ks], acc);
        }
        const int mb = (lane & 16) ? 8 : 0;
        #pragma unroll
        for (int j = 0; j < 8; ++j)
            pre[(size_t)(m0 + mb + j) * NP + n] = (_Float16)acc[j];
    }
}

// ---------------- sequential LSTM scan ----------------
// One workgroup per direction (grid=2). h kept in LDS (f16, A-layout-friendly),
// gates staged in LDS, cell state in registers. 2 barriers/step, 1000 steps.
// Job loop fully unrolled into 6 wave-uniform-guarded slots so the scheduler can
// software-pipeline tile j+1's loads under tile j's WMMAs.
__global__ void k_scan(const _Float16* __restrict__ preF, const _Float16* __restrict__ preB,
                       const _Float16* __restrict__ whhF, const _Float16* __restrict__ whhB,
                       _Float16* __restrict__ hF, _Float16* __restrict__ hB) {
    const int dir = blockIdx.x;
    const _Float16* pre = dir ? preB : preF;
    const _Float16* wpk = dir ? whhB : whhF;
    _Float16*      hout = dir ? hB : hF;

    __shared__ _Float16 hS[32 * KP];   // 12288 B : h_t as [B=32][K=192] f16, pads zero
    __shared__ _Float16 gS[32 * NP];   // 41984 B : gate pre-activations f16

    const int tid = threadIdx.x;       // 512 threads = 16 waves
    for (int e = tid; e < 32 * KP; e += 512) hS[e] = (_Float16)0.0f;
    float creg[11];
    #pragma unroll
    for (int j = 0; j < 11; ++j) creg[j] = 0.0f;
    __syncthreads();

    const int wave = tid >> 5, lane = tid & 31;
    for (int s = 0; s < T; ++s) {
        const int t = dir ? (T - 1 - s) : s;
        // -------- recurrent GEMM: gates = h @ w_hh^T + pre[t] --------
        #pragma unroll
        for (int jj = 0; jj < 6; ++jj) {
            const int job = wave + jj * 16;       // wave-uniform guard: EXEC all-ones
            if (job < 2 * NT_LSTM) {
                const int mt = job / NT_LSTM;     // batch tile 0..1
                const int nt = job - mt * NT_LSTM;
                // batch-preload B fragments (time-invariant, L2-hot) in one clause
                v16h bf[KS_LSTM];
                #pragma unroll
                for (int ks = 0; ks < KS_LSTM; ++ks)
                    bf[ks] = frag_b_packed(wpk, nt, ks, KS_LSTM, lane);
                const int n  = nt * 16 + (lane & 15);
                const int mb = mt * 16 + ((lane & 16) ? 8 : 0);
                // preload pre-activation addend so its latency overlaps the WMMAs
                float p[8];
                #pragma unroll
                for (int j = 0; j < 8; ++j)
                    p[j] = (float)pre[((size_t)(mb + j) * T + t) * NP + n];
                v8f acc = {};
                const _Float16* at = hS + mt * 16 * KP;
                #pragma unroll
                for (int ks = 0; ks < KS_LSTM; ++ks) {
                    v16h a = frag_a(at + ks * 32, KP, lane);
                    acc = wmma_f16(a, bf[ks], acc);
                }
                #pragma unroll
                for (int j = 0; j < 8; ++j)
                    gS[(mb + j) * NP + n] = (_Float16)(acc[j] + p[j]);
            }
        }
        __syncthreads();
        // -------- elementwise cell update (i,f,g,o gate order) --------
        int j = 0;
        for (int i = tid; i < B * H; i += 512, ++j) {
            const int bb = i / H;
            const int u  = i - bb * H;
            const float gi = (float)gS[bb * NP + u];
            const float gf = (float)gS[bb * NP + H + u];
            const float gg = (float)gS[bb * NP + 2 * H + u];
            const float go = (float)gS[bb * NP + 3 * H + u];
            const float cc = sigm(gf) * creg[j] + sigm(gi) * tanh_fast(gg);
            creg[j] = cc;
            const float hh = sigm(go) * tanh_fast(cc);
            const _Float16 h16 = (_Float16)hh;
            hS[bb * KP + u] = h16;
            hout[((size_t)bb * T + t) * HP + u] = h16;
        }
        __syncthreads();
    }
}

// ---------------- Linear(2H->H) + optional ReLU + residual ----------------
__global__ void k_linear_res(const _Float16* __restrict__ hF, const _Float16* __restrict__ hB,
                             const _Float16* __restrict__ WTp, const float* __restrict__ bl,
                             const float* __restrict__ xin, float* __restrict__ out,
                             int doRelu) {
    __shared__ _Float16 Atile[16 * K2P];  // 11264 B
    const int tid = threadIdx.x;          // 256 threads = 8 waves
    const long m0 = (long)blockIdx.x * 16;
    for (int e = tid; e < 16 * K2P; e += 256) {
        int r = e / K2P, k = e - r * K2P;
        size_t row = (size_t)(m0 + r);
        _Float16 v = (k < HP) ? hF[row * HP + k] : hB[row * HP + (k - HP)];
        Atile[e] = v;
    }
    __syncthreads();
    const int wave = tid >> 5, lane = tid & 31;
    for (int nt = wave; nt < NT_LIN; nt += 8) {
        v16h bf[KS_LIN];
        #pragma unroll
        for (int ks = 0; ks < KS_LIN; ++ks)
            bf[ks] = frag_b_packed(WTp, nt, ks, KS_LIN, lane);
        v8f acc = {};
        #pragma unroll
        for (int ks = 0; ks < KS_LIN; ++ks) {
            v16h a = frag_a(Atile + ks * 32, K2P, lane);
            acc = wmma_f16(a, bf[ks], acc);
        }
        const int n = nt * 16 + (lane & 15);
        if (n < H) {
            const float bias = bl[n];
            const int mb = (lane & 16) ? 8 : 0;
            #pragma unroll
            for (int j = 0; j < 8; ++j) {
                size_t row = (size_t)(m0 + mb + j);
                float v = acc[j] + bias;
                if (doRelu) v = fmaxf(v, 0.0f);
                out[row * H + n] = v + xin[row * H + n];
            }
        }
    }
}

// ---------------- host orchestration ----------------
extern "C" void kernel_launch(void* const* d_in, const int* in_sizes, int n_in,
                              void* d_out, int out_size, void* d_ws, size_t ws_size,
                              hipStream_t stream) {
    char* ws = (char*)d_ws;
    auto wih = [&](int blk, int dir) { return (_Float16*)(ws + OFF_WIH + (blk * 2 + dir) * SZ_PK); };
    auto whh = [&](int blk, int dir) { return (_Float16*)(ws + OFF_WHH + (blk * 2 + dir) * SZ_PK); };
    auto bsm = [&](int blk, int dir) { return (float*)   (ws + OFF_BS  + (blk * 2 + dir) * SZ_BS); };
    auto wtp = [&](int blk)          { return (_Float16*)(ws + OFF_WT  + blk * SZ_WT); };
    _Float16* preF = (_Float16*)(ws + OFF_PREF);
    _Float16* preB = (_Float16*)(ws + OFF_PREB);
    _Float16* hF   = (_Float16*)(ws + OFF_HF);
    _Float16* hB   = (_Float16*)(ws + OFF_HB);
    float*    ioA  = (float*)(ws + OFF_IOA);
    float*    ioB  = (float*)(ws + OFF_IOB);
    const float* x = (const float*)d_in[0];

    // ---- weight / bias prep ----
    const int packGrid = (int)((PK_LSTM + 255) / 256);
    const int linGrid  = (int)((PK_LIN + 255) / 256);
    for (int blk = 0; blk < 3; ++blk) {
        const int base = 1 + blk * 10;
        for (int dir = 0; dir < 2; ++dir) {
            const float* w_ih = (const float*)d_in[base + (dir ? 4 : 0)];
            const float* w_hh = (const float*)d_in[base + (dir ? 5 : 1)];
            const float* b_ih = (const float*)d_in[base + (dir ? 6 : 2)];
            const float* b_hh = (const float*)d_in[base + (dir ? 7 : 3)];
            k_pack_b<<<packGrid, 256, 0, stream>>>(w_ih, G4, H, wih(blk, dir), NT_LSTM, KS_LSTM);
            k_pack_b<<<packGrid, 256, 0, stream>>>(w_hh, G4, H, whh(blk, dir), NT_LSTM, KS_LSTM);
            k_bias_sum<<<3, 256, 0, stream>>>(b_ih, b_hh, bsm(blk, dir));
        }
        k_pack_b_lin<<<linGrid, 256, 0, stream>>>((const float*)d_in[base + 8], wtp(blk));
    }
    // zero hF+hB (contiguous) so padded columns are zero for the linear GEMM
    {
        long nwords = (long)(2 * SZ_H16 / 4);
        int zg = (int)((nwords + 255) / 256);
        k_zero32<<<zg, 256, 0, stream>>>((unsigned int*)hF, nwords);
    }

    // ---- three residual bi-LSTM blocks (stream-ordered) ----
    const int mGrid = (int)(BT / 16);   // 2000
    for (int blk = 0; blk < 3; ++blk) {
        const int base = 1 + blk * 10;
        const float* in  = (blk == 0) ? x : ((blk == 1) ? ioA : ioB);
        float*       oup = (blk == 2) ? (float*)d_out : ((blk == 0) ? ioA : ioB);
        const float* bl  = (const float*)d_in[base + 9];

        k_pregemm<<<mGrid, 256, 0, stream>>>(in, wih(blk, 0), wih(blk, 1),
                                             bsm(blk, 0), bsm(blk, 1), preF, preB);
        k_scan<<<2, 512, 0, stream>>>(preF, preB, whh(blk, 0), whh(blk, 1), hF, hB);
        k_linear_res<<<mGrid, 256, 0, stream>>>(hF, hB, wtp(blk), bl, in, oup,
                                                (blk < 2) ? 1 : 0);
    }
}
